// SimpleAttentionWithQKClip_17411797418096
// MI455X (gfx1250) — compile-verified, matrix operand order checked by reading, not run on hardware
//
#include <hip/hip_runtime.h>

#define DIM    2048
#define L_SEQ  2048
#define BATCH  2
#define NHEADS 16
#define HDIM   128
#define MROWS  (BATCH * L_SEQ)   // 4096

typedef __attribute__((ext_vector_type(16))) __bf16 v16bf;
typedef __attribute__((ext_vector_type(8)))  float  v8f;
typedef __attribute__((ext_vector_type(4)))  unsigned int u32x4;
typedef __attribute__((ext_vector_type(8)))  int i32x8;
typedef __attribute__((ext_vector_type(4)))  int i32x4;

// ---- fragment loads: lane half hf, element e -> K = k0 + e + 8*hf + (e>=8 ? 8 : 0)
// => two contiguous 16-byte loads. Works for global or LDS pointers (addrspace inferred).
__device__ __forceinline__ v16bf load_frag16(const __bf16* rowp, int k0, int hf) {
  union { v16bf v; uint4 u[2]; } f;
  f.u[0] = *reinterpret_cast<const uint4*>(rowp + k0 + 8 * hf);
  f.u[1] = *reinterpret_cast<const uint4*>(rowp + k0 + 16 + 8 * hf);
  return f.v;
}

__device__ __forceinline__ v8f wmma_bf16(v16bf a, v16bf b, v8f c) {
  return __builtin_amdgcn_wmma_f32_16x16x32_bf16(false, a, false, b, (short)0, c, false, false);
}

// ---- Tensor Data Mover: 2-D tile (tile_w x tile_h, 2-byte elems) global -> LDS.
// D# built per CDNA5 ISA ch.8: group0 {count=1, lds_addr, global_addr, type=2},
// group1 {data_size=2B, tensor dims, tile dims, dim0 stride}; groups 2/3 zero (2-D).
// This toolchain exposes the 6-arg builtin (g0, g1, g2, g3, extra group slot, cpol).
__device__ __forceinline__ void tdm_load_2d(const void* lds_ptr, const void* gptr,
                                            unsigned tile_w, unsigned tile_h,
                                            unsigned long long stride_elems) {
  unsigned long long ga = (unsigned long long)(size_t)gptr;
  unsigned lds_off = (unsigned)(size_t)lds_ptr;   // generic LDS addr low 32 = LDS offset
  u32x4 g0;
  g0.x = 1u;                                      // count=1, user mode, no gather
  g0.y = lds_off;                                 // lds_addr
  g0.z = (unsigned)ga;                            // global_addr[31:0]
  g0.w = (unsigned)((ga >> 32) & 0x01FFFFFFull) | (2u << 30);  // addr[56:32] | type=2
  const unsigned dim0 = (unsigned)stride_elems;   // row length (>= tile_w)
  const unsigned dim1 = 1u << 24;                 // generous row count (tiles always in-bounds)
  i32x8 g1;
  g1[0] = (int)(1u << 16);                                   // wg_mask=0, data_size=1 (2B)
  g1[1] = (int)((dim0 & 0xFFFFu) << 16);                     // tensor_dim0[15:0]
  g1[2] = (int)((dim0 >> 16) | ((dim1 & 0xFFFFu) << 16));    // dim0[31:16] | dim1[15:0]
  g1[3] = (int)((dim1 >> 16) | (tile_w << 16));              // dim1[31:16] | tile_dim0
  g1[4] = (int)(tile_h & 0xFFFFu);                           // tile_dim1, tile_dim2=0
  g1[5] = (int)(stride_elems & 0xFFFFFFFFull);               // tensor_dim0_stride[31:0]
  g1[6] = (int)((stride_elems >> 32) & 0xFFFFull);           // stride[47:32], dim1_stride=0
  g1[7] = 0;
  i32x4 z4 = {0, 0, 0, 0};
  i32x8 z8 = {0, 0, 0, 0, 0, 0, 0, 0};
  __builtin_amdgcn_tensor_load_to_lds(g0, g1, z4, z4, z8, 0);
}

__global__ void f32_to_bf16(const float* __restrict__ in, __bf16* __restrict__ out, int n) {
  int i = blockIdx.x * blockDim.x + threadIdx.x;
  int stride = gridDim.x * blockDim.x;
  for (; i < n; i += stride) out[i] = (__bf16)in[i];
}

// C = X * W^T + bias.  Block = 4 waves, block tile 128M x 64N; W tile staged in LDS by TDM,
// double-buffered. Each wave: 32M x 64N (8 accumulators), 8 WMMA per k-step.
// mode 0: Q [B,H,L,Hd]*scale   mode 1: K [B,H,L,Hd]   mode 2: V^T [B,H,Hd,L]
__global__ void __launch_bounds__(128) qkv_gemm(const __bf16* __restrict__ X,
                                                const __bf16* __restrict__ W,
                                                const float*  __restrict__ bias,
                                                __bf16* __restrict__ out,
                                                int mode, float scale) {
  __shared__ __bf16 sW[2][64 * 32];               // 4 KB per buffer
  const int lane = threadIdx.x & 31;
  const int wid  = threadIdx.x >> 5;
  const int hf   = lane >> 4;
  const int lr   = lane & 15;
  const int n0   = blockIdx.x * 64;
  const int m0   = blockIdx.y * 128 + wid * 32;

  v8f acc[2][4];
#pragma unroll
  for (int s = 0; s < 2; ++s)
#pragma unroll
    for (int t = 0; t < 4; ++t) acc[s][t] = v8f{};

  const __bf16* arow0 = X + (size_t)(m0 + lr) * DIM;
  const __bf16* arow1 = X + (size_t)(m0 + 16 + lr) * DIM;

  if (wid == 0) tdm_load_2d(&sW[0][0], W + (size_t)n0 * DIM, 32, 64, DIM);

  const int NK = DIM / 32;
  for (int ks = 0; ks < NK; ++ks) {
    const int k = ks * 32;
    if (wid == 0) {
      if (ks + 1 < NK) {
        tdm_load_2d(&sW[(ks + 1) & 1][0], W + (size_t)n0 * DIM + (k + 32), 32, 64, DIM);
        __builtin_amdgcn_s_wait_tensorcnt(1);     // current buffer resident
      } else {
        __builtin_amdgcn_s_wait_tensorcnt(0);
      }
    }
    __syncthreads();
    const __bf16* wb = &sW[ks & 1][0];
    v16bf a0 = load_frag16(arow0, k, hf);
    v16bf a1 = load_frag16(arow1, k, hf);
#pragma unroll
    for (int t = 0; t < 4; ++t) {
      v16bf b = load_frag16(wb + (t * 16 + lr) * 32, 0, hf);   // LDS row stride 32
      acc[0][t] = wmma_bf16(a0, b, acc[0][t]);
      acc[1][t] = wmma_bf16(a1, b, acc[1][t]);
    }
    __syncthreads();                              // next TDM may overwrite other buffer
  }

#pragma unroll
  for (int ms = 0; ms < 2; ++ms) {
    const int mbase = m0 + ms * 16;
    const int mb = mbase >> 11;                   // batch (L_SEQ == 2048)
    const int ml = mbase & (L_SEQ - 1);
#pragma unroll
    for (int t = 0; t < 4; ++t) {
      const int n  = n0 + t * 16 + lr;
      const float bs = bias[n];
      const int h = n >> 7, d = n & 127;
      const size_t bh = (size_t)(mb * NHEADS + h);
      if (mode == 2) {
        union { uint4 u; __bf16 e[8]; } pk;
#pragma unroll
        for (int r = 0; r < 8; ++r) pk.e[r] = (__bf16)(acc[ms][t][r] + bs);
        *reinterpret_cast<uint4*>(out + (bh * HDIM + d) * L_SEQ + ml + 8 * hf) = pk.u;
      } else {
#pragma unroll
        for (int r = 0; r < 8; ++r) {
          const int lq = ml + r + 8 * hf;
          out[(bh * L_SEQ + lq) * HDIM + d] = (__bf16)((acc[ms][t][r] + bs) * scale);
        }
      }
    }
  }
}

// Flash attention, S^T = K*Q^T formulation (per-lane scalar softmax state; the S^T
// accumulator is bit-identical to the B-fragment of O^T = V^T * P^T).
// K and V^T chunks staged in LDS by TDM, double-buffered, shared by all 4 waves.
__global__ void __launch_bounds__(128) attn(const __bf16* __restrict__ Q,
                                            const __bf16* __restrict__ K,
                                            const __bf16* __restrict__ Vt,
                                            float* __restrict__ out) {
  __shared__ __bf16 sK[2][32 * HDIM];             // 8 KB per buffer
  __shared__ __bf16 sV[2][HDIM * 32];             // 8 KB per buffer
  const int lane = threadIdx.x & 31;
  const int wid  = threadIdx.x >> 5;
  const int hf   = lane >> 4;
  const int lr   = lane & 15;
  const int bh   = blockIdx.y;
  const int q0   = blockIdx.x * 64 + wid * 16;

  const __bf16* qrow = Q + ((size_t)bh * L_SEQ + q0 + lr) * HDIM;
  v16bf qf[4];
#pragma unroll
  for (int s = 0; s < 4; ++s) qf[s] = load_frag16(qrow, s * 32, hf);

  const __bf16* kg = K  + (size_t)bh * L_SEQ * HDIM;
  const __bf16* vg = Vt + (size_t)bh * HDIM * L_SEQ;

  if (wid == 0) {
    tdm_load_2d(&sK[0][0], kg, HDIM, 32, HDIM);   // 32 key rows, contiguous
    tdm_load_2d(&sV[0][0], vg, 32, HDIM, L_SEQ);  // 128 d-rows x 32 keys, stride L
  }

  v8f accO[8];
#pragma unroll
  for (int t = 0; t < 8; ++t) accO[t] = v8f{};
  float rmax = -1.0e30f, rsum = 0.0f;

  const int NC = L_SEQ / 32;
  for (int c = 0; c < NC; ++c) {
    const int kc = c * 32;
    if (wid == 0) {
      if (c + 1 < NC) {
        tdm_load_2d(&sK[(c + 1) & 1][0], kg + (size_t)(kc + 32) * HDIM, HDIM, 32, HDIM);
        tdm_load_2d(&sV[(c + 1) & 1][0], vg + (kc + 32), 32, HDIM, L_SEQ);
        __builtin_amdgcn_s_wait_tensorcnt(2);     // current K/V pair resident
      } else {
        __builtin_amdgcn_s_wait_tensorcnt(0);
      }
    }
    __syncthreads();
    const __bf16* kb = &sK[c & 1][0];
    const __bf16* vb = &sV[c & 1][0];

    // scores: two 16-key S^T tiles over 4 d-steps
    v8f s0 = v8f{}, s1 = v8f{};
#pragma unroll
    for (int s = 0; s < 4; ++s) {
      v16bf a0 = load_frag16(kb + lr * HDIM, s * 32, hf);
      v16bf a1 = load_frag16(kb + (16 + lr) * HDIM, s * 32, hf);
      s0 = wmma_bf16(a0, qf[s], s0);
      s1 = wmma_bf16(a1, qf[s], s1);
    }
    // online softmax: per-lane scalars, one cross-half shuffle per reduction
    float cmax = -1.0e30f;
#pragma unroll
    for (int r = 0; r < 8; ++r) { cmax = fmaxf(cmax, s0[r]); cmax = fmaxf(cmax, s1[r]); }
    cmax = fmaxf(cmax, __shfl_xor(cmax, 16, 32));
    const float nmax = fmaxf(rmax, cmax);
    const float corr = __expf(rmax - nmax);
    float p0[8], p1[8], psum = 0.0f;
#pragma unroll
    for (int r = 0; r < 8; ++r) {
      p0[r] = __expf(s0[r] - nmax);
      p1[r] = __expf(s1[r] - nmax);
      psum += p0[r] + p1[r];
    }
    psum += __shfl_xor(psum, 16, 32);
    rsum = rsum * corr + psum;
    rmax = nmax;
    union { v16bf v; __bf16 e[16]; } pf;       // P^T C-layout == PV B-fragment layout
#pragma unroll
    for (int r = 0; r < 8; ++r) { pf.e[r] = (__bf16)p0[r]; pf.e[r + 8] = (__bf16)p1[r]; }
#pragma unroll
    for (int dt = 0; dt < 8; ++dt) {
      v16bf vf = load_frag16(vb + (dt * 16 + lr) * 32, 0, hf);
      accO[dt] = accO[dt] * corr;
      accO[dt] = wmma_bf16(vf, pf.v, accO[dt]);
    }
    __syncthreads();
  }

  const float inv = 1.0f / rsum;
  const int b = bh >> 4, h = bh & 15;
  float* orow = out + ((size_t)b * L_SEQ + (q0 + lr)) * DIM + h * HDIM;
#pragma unroll
  for (int dt = 0; dt < 8; ++dt)
#pragma unroll
    for (int r = 0; r < 8; ++r)
      orow[dt * 16 + r + 8 * hf] = accO[dt][r] * inv;
}

extern "C" void kernel_launch(void* const* d_in, const int* in_sizes, int n_in,
                              void* d_out, int out_size, void* d_ws, size_t ws_size,
                              hipStream_t stream) {
  const float* x  = (const float*)d_in[0];
  const float* Wq = (const float*)d_in[1];
  const float* bq = (const float*)d_in[2];
  const float* Wk = (const float*)d_in[3];
  const float* bk = (const float*)d_in[4];
  const float* Wv = (const float*)d_in[5];
  const float* bv = (const float*)d_in[6];

  const size_t XN = (size_t)MROWS * DIM;
  const size_t WN = (size_t)DIM * DIM;
  __bf16* xb  = (__bf16*)d_ws;
  __bf16* wqb = xb  + XN;
  __bf16* wkb = wqb + WN;
  __bf16* wvb = wkb + WN;
  __bf16* qb  = wvb + WN;
  __bf16* kb  = qb  + XN;
  __bf16* vtb = kb  + XN;   // ~92 MB workspace total

  f32_to_bf16<<<2048, 256, 0, stream>>>(x,  xb,  (int)XN);
  f32_to_bf16<<<1024, 256, 0, stream>>>(Wq, wqb, (int)WN);
  f32_to_bf16<<<1024, 256, 0, stream>>>(Wk, wkb, (int)WN);
  f32_to_bf16<<<1024, 256, 0, stream>>>(Wv, wvb, (int)WN);

  dim3 gg(DIM / 64, MROWS / 128);
  const float scale = 0.088388347648318447f;  // 1/sqrt(128), folded into Q
  qkv_gemm<<<gg, 128, 0, stream>>>(xb, wqb, bq, qb,  0, scale);
  qkv_gemm<<<gg, 128, 0, stream>>>(xb, wkb, bk, kb,  1, 1.0f);
  qkv_gemm<<<gg, 128, 0, stream>>>(xb, wvb, bv, vtb, 2, 1.0f);

  dim3 ag(L_SEQ / 64, BATCH * NHEADS);
  attn<<<ag, 128, 0, stream>>>(qb, kb, vtb, (float*)d_out);
}
